// MC_FRGE_49254684950667
// MI455X (gfx1250) — matrix operand order repaired and checked by reading, not compiled, not run on hardware
//
#include <hip/hip_runtime.h>

// ---------------------------------------------------------------------------
// MI455X (gfx1250, wave32): dense contractions on v_wmma_f32_16x16x32_f16
// (f32 accumulate), 32x32 macro-tile per wave (2x2 WMMA, fragment reuse).
// GEMM templated on (N,K) so strides/guards constant-fold into load offsets.
// Irregular GAT edge softmax/aggregation -> VALU + global atomics.
// ---------------------------------------------------------------------------

typedef __attribute__((ext_vector_type(16))) _Float16 v16h;
typedef __attribute__((ext_vector_type(8)))  _Float16 v8h;
typedef __attribute__((ext_vector_type(8)))  float    v8f;

#define WMMA16(a, b, c) \
  __builtin_amdgcn_wmma_f32_16x16x32_f16(false, (a), false, (b), (short)0, (c), false, false)

namespace {
constexpr int Bn = 64, Sn = 196, Rn = 512, Vv = 2000, Dn = 256;
constexpr int Kg = 5, Cn = 10, Hn = 4, DHn = 64;
constexpr int SPAD = 224;               // S=196 padded to 7 k-steps of 32
}

// ----------------------------- fragment helpers ----------------------------

__device__ inline v16h cat8(v8h lo, v8h hi) {
  return __builtin_shufflevector(lo, hi, 0, 1, 2, 3, 4, 5, 6, 7,
                                 8, 9, 10, 11, 12, 13, 14, 15);
}

__device__ inline v8h cvt8(float4 a, float4 b) {
  v8h r;
  r[0] = (_Float16)a.x; r[1] = (_Float16)a.y; r[2] = (_Float16)a.z; r[3] = (_Float16)a.w;
  r[4] = (_Float16)b.x; r[5] = (_Float16)b.y; r[6] = (_Float16)b.z; r[7] = (_Float16)b.w;
  return r;
}

__device__ inline v16h scale16(v16h v, _Float16 s) {
#pragma unroll
  for (int e = 0; e < 16; ++e) v[e] = v[e] * s;
  return v;
}

// A-frag: rows contiguous in k -> 4x global_load_b128 from one base.
// base points at A[ma*lda + k0 + half*8].
__device__ inline v16h load_a_fast(const float* __restrict__ A, int base) {
  const float4* p0 = (const float4*)(A + base);
  const float4* p1 = (const float4*)(A + base + 16);
  return cat8(cvt8(p0[0], p0[1]), cvt8(p1[0], p1[1]));
}

// B-frag: one base pointer + compile-time immediate offsets (((e&7) + 16*(e>>3))*LDB).
template<int LDB>
__device__ inline v16h load_b_fast(const float* __restrict__ Bm, int k0h, int nb) {
  const float* p = Bm + k0h * LDB + nb;
  v16h r;
#pragma unroll
  for (int e = 0; e < 16; ++e) r[e] = (_Float16)p[((e & 7) + ((e >> 3) << 4)) * LDB];
  return r;
}

// K-remainder masked variants (zero-fill beyond K; OOB rows may hold NaN bits).
template<int K_>
__device__ inline v16h load_a_kmask(const float* __restrict__ A, int ma, int k0, int half) {
  v16h r;
#pragma unroll
  for (int e = 0; e < 16; ++e) {
    int k = k0 + (e & 7) + half * 8 + ((e >> 3) << 4);
    bool ok = k < K_;
    float v = A[ma * K_ + (ok ? k : 0)];
    r[e] = (_Float16)(ok ? v : 0.0f);
  }
  return r;
}

template<int LDB, int K_>
__device__ inline v16h load_b_kmask(const float* __restrict__ Bm, int k0h, int nb) {
  v16h r;
#pragma unroll
  for (int e = 0; e < 16; ++e) {
    int k = k0h + (e & 7) + ((e >> 3) << 4);
    bool ok = k < K_;
    float v = Bm[(ok ? k : 0) * LDB + nb];
    r[e] = (_Float16)(ok ? v : 0.0f);
  }
  return r;
}

template<int N_>
__device__ inline void store_tile(float* __restrict__ Cm, int mm0, int nn0,
                                  int half, int l, v8f acc,
                                  const float* __restrict__ bias, int relu) {
  const int n = nn0 + l;
  float bv = bias ? bias[n] : 0.0f;     // one load; same column for all rows
  float* p = Cm + (mm0 + half * 8) * N_ + n;
#pragma unroll
  for (int r = 0; r < 8; ++r) {
    float v = acc[r] + bv;
    if (relu) v = fmaxf(v, 0.0f);
    p[r * N_] = v;
  }
}

// ---------------------------------------------------------------------------
// WMMA GEMM: C[M,N_] (+)= A[M,K_] @ B[K_,N_]; one wave per 32x32 macro-tile
// (2x2 v_wmma, fragment reuse). M must be a multiple of 32 (true for all
// call sites). N-edge tiles (N_=40) use column clamping: WMMA columns are
// independent, so clamped-column garbage only lands in unstored lanes.
// ---------------------------------------------------------------------------
template<int N_, int K_>
__global__ __launch_bounds__(128)
void gemm_wmma(const float* __restrict__ A, const float* __restrict__ Bm,
               float* __restrict__ Cm, int M,
               const float* __restrict__ bias,
               const float* __restrict__ rowscale,   // mask rows, period Rn
               int accumulate, int relu)
{
  constexpr int Kfull = K_ & ~31;
  constexpr int nt32 = (N_ + 31) >> 5;
  const int lane = threadIdx.x & 31;
  const int tile = blockIdx.x * (blockDim.x >> 5) + (threadIdx.x >> 5);
  const int mt32 = M >> 5;
  if (tile >= mt32 * nt32) return;       // wave-uniform
  const int mt = tile / nt32, nt = tile % nt32;
  const int m0 = mt << 5, n0 = nt << 5;
  const int half = lane >> 4, l = lane & 15;
  const bool interior = (N_ % 32 == 0) || (n0 + 32 <= N_);   // wave-uniform

  if (interior) {
    const int ma0 = m0 + l, ma1 = m0 + 16 + l;
    const int nb0 = n0 + l, nb1 = n0 + 16 + l;
    _Float16 rs0 = (_Float16)1.0f, rs1 = (_Float16)1.0f;
    if (rowscale) {
      rs0 = (_Float16)rowscale[ma0 & (Rn - 1)];
      rs1 = (_Float16)rowscale[ma1 & (Rn - 1)];
    }
    v8f a00 = {}, a01 = {}, a10 = {}, a11 = {};
    if (accumulate) {
      const float* p0 = Cm + (m0 + half * 8) * N_;
      const float* p1 = Cm + (m0 + 16 + half * 8) * N_;
#pragma unroll
      for (int r = 0; r < 8; ++r) {
        a00[r] = p0[r * N_ + nb0]; a01[r] = p0[r * N_ + nb1];
        a10[r] = p1[r * N_ + nb0]; a11[r] = p1[r * N_ + nb1];
      }
    }
    for (int k0 = 0; k0 < Kfull; k0 += 32) {
      if (k0 + 32 < Kfull)               // gfx1250 global_prefetch_b8
        __builtin_prefetch(&Bm[(k0 + 32 + half * 16) * N_ + nb0], 0, 3);
      v16h af0 = load_a_fast(A, ma0 * K_ + k0 + half * 8);
      v16h af1 = load_a_fast(A, ma1 * K_ + k0 + half * 8);
      if (rowscale) { af0 = scale16(af0, rs0); af1 = scale16(af1, rs1); }
      v16h bf0 = load_b_fast<N_>(Bm, k0 + half * 8, nb0);
      v16h bf1 = load_b_fast<N_>(Bm, k0 + half * 8, nb1);
      a00 = WMMA16(af0, bf0, a00);
      a01 = WMMA16(af0, bf1, a01);
      a10 = WMMA16(af1, bf0, a10);
      a11 = WMMA16(af1, bf1, a11);
    }
    if constexpr (Kfull < K_) {          // K remainder (K_=2000 only)
      v16h af0 = load_a_kmask<K_>(A, ma0, Kfull, half);
      v16h af1 = load_a_kmask<K_>(A, ma1, Kfull, half);
      if (rowscale) { af0 = scale16(af0, rs0); af1 = scale16(af1, rs1); }
      v16h bf0 = load_b_kmask<N_, K_>(Bm, Kfull + half * 8, nb0);
      v16h bf1 = load_b_kmask<N_, K_>(Bm, Kfull + half * 8, nb1);
      a00 = WMMA16(af0, bf0, a00);
      a01 = WMMA16(af0, bf1, a01);
      a10 = WMMA16(af1, bf0, a10);
      a11 = WMMA16(af1, bf1, a11);
    }
    store_tile<N_>(Cm, m0,      n0,      half, l, a00, bias, relu);
    store_tile<N_>(Cm, m0,      n0 + 16, half, l, a01, bias, relu);
    store_tile<N_>(Cm, m0 + 16, n0,      half, l, a10, bias, relu);
    store_tile<N_>(Cm, m0 + 16, n0 + 16, half, l, a11, bias, relu);
  } else if constexpr (N_ % 32 != 0) {
    // N-edge macro-tile: clamp columns, guard stores only.
#pragma unroll
    for (int sm = 0; sm < 2; ++sm) {
      int mm0 = m0 + sm * 16;
#pragma unroll
      for (int sn = 0; sn < 2; ++sn) {
        int nn0 = n0 + sn * 16;
        if (nn0 >= N_) continue;         // wave-uniform
        const int ma = mm0 + l;
        const int nb = nn0 + l;
        const int nbc = nb < N_ ? nb : N_ - 1;   // clamped column
        _Float16 rs = (_Float16)1.0f;
        if (rowscale) rs = (_Float16)rowscale[ma & (Rn - 1)];
        v8f acc = {};
        if (accumulate) {
          const float* p = Cm + (mm0 + half * 8) * N_ + nbc;
#pragma unroll
          for (int r = 0; r < 8; ++r) acc[r] = p[r * N_];
        }
        for (int k0 = 0; k0 < Kfull; k0 += 32) {
          v16h af = load_a_fast(A, ma * K_ + k0 + half * 8);
          if (rowscale) af = scale16(af, rs);
          v16h bf = load_b_fast<N_>(Bm, k0 + half * 8, nbc);
          acc = WMMA16(af, bf, acc);
        }
        if constexpr (Kfull < K_) {
          v16h af = load_a_kmask<K_>(A, ma, Kfull, half);
          if (rowscale) af = scale16(af, rs);
          v16h bf = load_b_kmask<N_, K_>(Bm, Kfull + half * 8, nbc);
          acc = WMMA16(af, bf, acc);
        }
        if (nb < N_) {                   // divergent store guard (edge only)
          float bv = bias ? bias[nb] : 0.0f;
          float* p = Cm + (mm0 + half * 8) * N_ + nb;
#pragma unroll
          for (int r = 0; r < 8; ++r) {
            float v = acc[r] + bv;
            if (relu) v = fmaxf(v, 0.0f);
            p[r * N_] = v;
          }
        }
      }
    }
  }
}

// ---------------------------------------------------------------------------
// Cross-attention, one wave per (16-row tile, head, batch).
// scores = (Q Kk^T)/8 -> LDS f32; softmax over S -> f16 probs in LDS;
// out = P @ V: ks outer (prob A-frag as 2x ds_load_b128, loaded once),
// 4 n-tiles inner with 4 live accumulators. Out-of-range s rows are
// index-clamped; their probs are exactly 0, so products vanish.
// ---------------------------------------------------------------------------
__global__ __launch_bounds__(32)
void attn_kernel(const float* __restrict__ q, const float* __restrict__ kk,
                 const float* __restrict__ vv, float* __restrict__ attn)
{
  __shared__ float    sc[16][SPAD + 8];
  __shared__ _Float16 pr[16][SPAD];
  const int lane = threadIdx.x, half = lane >> 4, l = lane & 15;
  const int rt = blockIdx.x, h = blockIdx.y, b = blockIdx.z;
  const float scale = 0.125f;            // 1/sqrt(64)
  constexpr int LD = Hn * DHn;           // 256

  v16h qa[2];
  const int m = rt * 16 + l;             // < R always
#pragma unroll
  for (int kb = 0; kb < 2; ++kb)
    qa[kb] = load_a_fast(q, m * LD + h * DHn + kb * 32 + half * 8);

  for (int st = 0; st < SPAD / 16; ++st) {
    v8f accs = {};
    const int s = st * 16 + l;
    const int se = s < Sn ? s : Sn - 1;  // clamp; junk columns ignored later
#pragma unroll
    for (int kb = 0; kb < 2; ++kb) {
      v16h bf = load_a_fast(kk, (b * Sn + se) * LD + h * DHn + kb * 32 + half * 8);
      accs = WMMA16(qa[kb], bf, accs);
    }
#pragma unroll
    for (int r = 0; r < 8; ++r)
      sc[r + half * 8][st * 16 + l] = accs[r] * scale;
  }
  __syncthreads();

  if (lane < 16) {                       // per-row softmax over S
    float mx = -1e30f;
    for (int c = 0; c < Sn; ++c) mx = fmaxf(mx, sc[lane][c]);
    float sum = 0.0f;
    for (int c = 0; c < Sn; ++c) sum += __expf(sc[lane][c] - mx);
    float inv = 1.0f / sum;
    for (int c = 0; c < SPAD; ++c)
      pr[lane][c] = (_Float16)((c < Sn) ? __expf(sc[lane][c] - mx) * inv : 0.0f);
  }
  __syncthreads();

  v8f acco[4] = {{}, {}, {}, {}};
  for (int ks = 0; ks < SPAD / 32; ++ks) {
    const int k0 = ks * 32;
    v8h lo = *(const v8h*)&pr[l][k0 + half * 8];        // ds_load_b128
    v8h hi = *(const v8h*)&pr[l][k0 + 16 + half * 8];   // ds_load_b128
    v16h af = cat8(lo, hi);
    const float* vrow[16];               // clamped row bases, reused x4
#pragma unroll
    for (int e = 0; e < 16; ++e) {
      int s = k0 + (e & 7) + half * 8 + ((e >> 3) << 4);
      int se = s < Sn ? s : Sn - 1;
      vrow[e] = vv + (b * Sn + se) * LD + h * DHn + l;
    }
#pragma unroll
    for (int nt = 0; nt < 4; ++nt) {
      v16h bf;
#pragma unroll
      for (int e = 0; e < 16; ++e) bf[e] = (_Float16)vrow[e][nt * 16];
      acco[nt] = WMMA16(af, bf, acco[nt]);
    }
  }
#pragma unroll
  for (int nt = 0; nt < 4; ++nt) {
    float* p = attn + (b * Rn + rt * 16 + half * 8) * LD + h * DHn + nt * 16 + l;
#pragma unroll
    for (int r = 0; r < 8; ++r) p[r * LD] = acco[nt][r];
  }
}

// ---------------------------------------------------------------------------
// GAT irregular part: per-node logits, edge softmax via atomics, scatter.
// ---------------------------------------------------------------------------
__device__ inline unsigned fenc(float x) {
  unsigned u = __float_as_uint(x);
  return (u & 0x80000000u) ? ~u : (u | 0x80000000u);
}
__device__ inline float fdec(unsigned u) {
  return (u & 0x80000000u) ? __uint_as_float(u & 0x7FFFFFFFu) : __uint_as_float(~u);
}
__device__ inline float leaky(float x) { return x > 0.0f ? x : 0.2f * x; }

__global__ void node_logits_kernel(const float* __restrict__ hbuf,
                                   const float* __restrict__ a_s,
                                   const float* __restrict__ a_d,
                                   float* __restrict__ es, float* __restrict__ ed, int F)
{
  int idx = blockIdx.x * blockDim.x + threadIdx.x;   // over B*R*H
  if (idx >= Bn * Rn * Hn) return;
  int h = idx % Hn, br = idx / Hn;
  const float* row = hbuf + (size_t)br * Hn * F + h * F;
  float s = 0.0f, d = 0.0f;
  for (int f = 0; f < F; ++f) { s += row[f] * a_s[h * F + f]; d += row[f] * a_d[h * F + f]; }
  es[idx] = s; ed[idx] = d;
}

__global__ void edge_max_kernel(const float* __restrict__ es, const float* __restrict__ ed,
                                const int* __restrict__ ei, int nE,
                                unsigned* __restrict__ menc)
{
  int idx = blockIdx.x * blockDim.x + threadIdx.x;   // over B*nE*H
  if (idx >= Bn * nE * Hn) return;
  int h = idx % Hn, t = idx / Hn, e = t % nE, b = t / nE;
  int s = ei[e], d = ei[nE + e];
  float x = leaky(es[(b * Rn + s) * Hn + h] + ed[(b * Rn + d) * Hn + h]);
  atomicMax(&menc[(b * Rn + d) * Hn + h], fenc(x));
}

__global__ void edge_denom_kernel(const float* __restrict__ es, const float* __restrict__ ed,
                                  const int* __restrict__ ei, int nE,
                                  const unsigned* __restrict__ menc,
                                  float* __restrict__ denom)
{
  int idx = blockIdx.x * blockDim.x + threadIdx.x;
  if (idx >= Bn * nE * Hn) return;
  int h = idx % Hn, t = idx / Hn, e = t % nE, b = t / nE;
  int s = ei[e], d = ei[nE + e];
  int di = (b * Rn + d) * Hn + h;
  float x = leaky(es[(b * Rn + s) * Hn + h] + ed[di]);
  atomicAdd(&denom[di], __expf(x - fdec(menc[di])));
}

__global__ void edge_scatter_kernel(const float* __restrict__ es, const float* __restrict__ ed,
                                    const int* __restrict__ ei, int nE,
                                    const unsigned* __restrict__ menc,
                                    const float* __restrict__ denom,
                                    const float* __restrict__ hbuf,
                                    float* __restrict__ gout, int F)
{
  int idx = blockIdx.x * blockDim.x + threadIdx.x;
  if (idx >= Bn * nE * Hn) return;
  int h = idx % Hn, t = idx / Hn, e = t % nE, b = t / nE;
  int s = ei[e], d = ei[nE + e];
  int di = (b * Rn + d) * Hn + h;
  float x = leaky(es[(b * Rn + s) * Hn + h] + ed[di]);
  float alpha = __expf(x - fdec(menc[di])) / (denom[di] + 1e-16f);
  const float* hs = hbuf + ((size_t)(b * Rn + s) * Hn + h) * F;
  float* go = gout + ((size_t)di) * F;
  for (int f = 0; f < F; ++f) atomicAdd(&go[f], hs[f] * alpha);
}

__global__ void head_mean_kernel(const float* __restrict__ gout,
                                 const float* __restrict__ bias,
                                 float* __restrict__ xout, int F, int relu)
{
  int idx = blockIdx.x * blockDim.x + threadIdx.x;   // over B*R*F
  if (idx >= Bn * Rn * F) return;
  int f = idx % F, br = idx / F;
  float s = 0.0f;
  for (int h = 0; h < Hn; ++h) s += gout[(size_t)br * Hn * F + h * F + f];
  float v = s * (1.0f / Hn) + bias[f];
  if (relu) v = fmaxf(v, 0.0f);
  xout[idx] = v;
}

__global__ void rule_sum_kernel(const float* __restrict__ x3, float* __restrict__ logits,
                                int kgroup)
{
  int idx = blockIdx.x * blockDim.x + threadIdx.x;   // over B*C
  if (idx >= Bn * Cn) return;
  int c = idx % Cn, b = idx / Cn;
  float s = 0.0f;
  for (int r = 0; r < Rn; ++r) s += x3[(size_t)(b * Rn + r) * Cn + c];
  logits[kgroup * Bn * Cn + idx] = s;
}

__global__ void logsoftmax_kernel(const float* __restrict__ logits, float* __restrict__ out)
{
  int idx = blockIdx.x * blockDim.x + threadIdx.x;   // over Kg*B
  if (idx >= Kg * Bn) return;
  const float* row = logits + idx * Cn;
  float mx = -1e30f;
  for (int c = 0; c < Cn; ++c) mx = fmaxf(mx, row[c]);
  float s = 0.0f;
  for (int c = 0; c < Cn; ++c) s += __expf(row[c] - mx);
  float lse = mx + __logf(s);
  for (int c = 0; c < Cn; ++c) out[idx * Cn + c] = row[c] - lse;
}

// ---------------------------------------------------------------------------
// Host orchestration
// ---------------------------------------------------------------------------
static inline int gemm_grid(int M, int N) {
  int tiles = (M / 32) * ((N + 31) / 32);
  return (tiles + 3) / 4;                 // 4 waves / block
}

extern "C" void kernel_launch(void* const* d_in, const int* in_sizes, int n_in,
                              void* d_out, int out_size, void* d_ws, size_t ws_size,
                              hipStream_t stream) {
  const float* vis = (const float*)d_in[0];
  const float* wb  = (const float*)d_in[1];  const float* bb = (const float*)d_in[2];
  const float* wk  = (const float*)d_in[3];  const float* bk = (const float*)d_in[4];
  const float* wq  = (const float*)d_in[5];  const float* wkk = (const float*)d_in[6];
  const float* wv  = (const float*)d_in[7];  const float* wo = (const float*)d_in[8];
  const float* bo  = (const float*)d_in[9];
  const float* g1W = (const float*)d_in[10]; const float* g1as = (const float*)d_in[11];
  const float* g1ad = (const float*)d_in[12]; const float* g1b = (const float*)d_in[13];
  const float* g2W = (const float*)d_in[14]; const float* g2as = (const float*)d_in[15];
  const float* g2ad = (const float*)d_in[16]; const float* g2b = (const float*)d_in[17];
  const float* g3W = (const float*)d_in[18]; const float* g3as = (const float*)d_in[19];
  const float* g3ad = (const float*)d_in[20]; const float* g3b = (const float*)d_in[21];
  const float* bmh = (const float*)d_in[22]; const float* cmh = (const float*)d_in[23];
  const float* mask = (const float*)d_in[24];
  const int* ei = (const int*)d_in[25];
  const int nE = in_sizes[25] / 2;

  float* ws = (float*)d_ws;
  size_t o = 0;
  float* rule = ws + o; o += (size_t)Rn * Dn;
  float* qb   = ws + o; o += (size_t)Rn * Dn;
  float* kkb  = ws + o; o += (size_t)Bn * Sn * Dn;
  float* vvb  = ws + o; o += (size_t)Bn * Sn * Dn;
  float* attn = ws + o; o += (size_t)Bn * Rn * Dn;
  float* emb  = ws + o; o += (size_t)Bn * Rn * Dn;
  float* hbuf = attn;                           // attn dead after emb GEMM
  float* gout = ws + o; o += (size_t)Bn * Rn * Dn;
  float* x1   = ws + o; o += (size_t)Bn * Rn * 64;
  float* x2   = ws + o; o += (size_t)Bn * Rn * 32;
  float* x3   = ws + o; o += (size_t)Bn * Rn * Cn;
  float* es   = ws + o; o += (size_t)Bn * Rn * Hn;
  float* ed   = ws + o; o += (size_t)Bn * Rn * Hn;
  unsigned* menc = (unsigned*)(ws + o); o += (size_t)Bn * Rn * Hn;
  float* denom = ws + o; o += (size_t)Bn * Rn * Hn;
  float* logits = ws + o; o += (size_t)Kg * Bn * Cn;

  // (1) rule_emb = basic_mh@wb + bb + crucial_mh@wk + bk   [512x2000 @ 2000x256]
  gemm_wmma<Dn, Vv><<<gemm_grid(Rn, Dn), 128, 0, stream>>>(
      bmh, wb, rule, Rn, bb, nullptr, 0, 0);
  gemm_wmma<Dn, Vv><<<gemm_grid(Rn, Dn), 128, 0, stream>>>(
      cmh, wk, rule, Rn, bk, nullptr, 1, 0);

  // (2) QKV projections  [*x256 @ 256x256]
  gemm_wmma<Dn, Dn><<<gemm_grid(Rn, Dn), 128, 0, stream>>>(
      rule, wq, qb, Rn, nullptr, nullptr, 0, 0);
  gemm_wmma<Dn, Dn><<<gemm_grid(Bn * Sn, Dn), 128, 0, stream>>>(
      vis, wkk, kkb, Bn * Sn, nullptr, nullptr, 0, 0);
  gemm_wmma<Dn, Dn><<<gemm_grid(Bn * Sn, Dn), 128, 0, stream>>>(
      vis, wv, vvb, Bn * Sn, nullptr, nullptr, 0, 0);

  // (3) cross-attention
  attn_kernel<<<dim3(Rn / 16, Hn, Bn), 32, 0, stream>>>(qb, kkb, vvb, attn);

  // (4) output projection  [32768x256 @ 256x256]
  gemm_wmma<Dn, Dn><<<gemm_grid(Bn * Rn, Dn), 128, 0, stream>>>(
      attn, wo, emb, Bn * Rn, bo, nullptr, 0, 0);

  // (5) K grouped 3-layer GATs
  auto gat_rest = [&](const float* as_, const float* ad_, const float* bias,
                      float* xout, int F, int relu) {
    const int N = Hn * F;
    const int nBRH = Bn * Rn * Hn;
    node_logits_kernel<<<(nBRH + 255) / 256, 256, 0, stream>>>(hbuf, as_, ad_, es, ed, F);
    hipMemsetAsync(menc, 0, nBRH * sizeof(unsigned), stream);
    hipMemsetAsync(denom, 0, nBRH * sizeof(float), stream);
    hipMemsetAsync(gout, 0, (size_t)Bn * Rn * N * sizeof(float), stream);
    const int nBEH = Bn * nE * Hn;
    edge_max_kernel<<<(nBEH + 255) / 256, 256, 0, stream>>>(es, ed, ei, nE, menc);
    edge_denom_kernel<<<(nBEH + 255) / 256, 256, 0, stream>>>(es, ed, ei, nE, menc, denom);
    edge_scatter_kernel<<<(nBEH + 255) / 256, 256, 0, stream>>>(es, ed, ei, nE, menc, denom,
                                                                hbuf, gout, F);
    const int nBRF = Bn * Rn * F;
    head_mean_kernel<<<(nBRF + 255) / 256, 256, 0, stream>>>(gout, bias, xout, F, relu);
  };

  for (int i = 0; i < Kg; ++i) {
    // layer1: [32768x256 @ 256x256], masked rows
    gemm_wmma<256, 256><<<gemm_grid(Bn * Rn, 256), 128, 0, stream>>>(
        emb, g1W + (size_t)i * Dn * 256, hbuf, Bn * Rn, nullptr, mask + i * Rn, 0, 0);
    gat_rest(g1as + i * Hn * 64, g1ad + i * Hn * 64, g1b + i * 64, x1, 64, 1);
    // layer2: [32768x64 @ 64x128]
    gemm_wmma<128, 64><<<gemm_grid(Bn * Rn, 128), 128, 0, stream>>>(
        x1, g2W + (size_t)i * 64 * 128, hbuf, Bn * Rn, nullptr, nullptr, 0, 0);
    gat_rest(g2as + i * Hn * 32, g2ad + i * Hn * 32, g2b + i * 32, x2, 32, 1);
    // layer3: [32768x32 @ 32x40]
    gemm_wmma<40, 32><<<gemm_grid(Bn * Rn, 40), 128, 0, stream>>>(
        x2, g3W + (size_t)i * 32 * 40, hbuf, Bn * Rn, nullptr, nullptr, 0, 0);
    gat_rest(g3as + i * Hn * Cn, g3ad + i * Hn * Cn, g3b + i * Cn, x3, Cn, 0);
    rule_sum_kernel<<<(Bn * Cn + 255) / 256, 256, 0, stream>>>(x3, logits, i);
  }

  // (6) log-softmax over classes -> d_out [K,B,C]
  logsoftmax_kernel<<<(Kg * Bn + 255) / 256, 256, 0, stream>>>(logits, (float*)d_out);
}
